// GAT_65206193488150
// MI455X (gfx1250) — matrix operand order repaired and checked by reading, not compiled, not run on hardware
//
#include <hip/hip_runtime.h>
#include <hip/hip_bf16.h>

// ---------------------------------------------------------------------------
// Types for WMMA
// ---------------------------------------------------------------------------
typedef __attribute__((ext_vector_type(16))) __bf16          v16bf;
typedef __attribute__((ext_vector_type(16))) unsigned short  v16us;
typedef __attribute__((ext_vector_type(8)))  unsigned short  v8us;
typedef __attribute__((ext_vector_type(8)))  float           v8f;

#define HC    256   // H*C
#define HEADS 4
#define CH    64

__device__ __forceinline__ unsigned short f32_to_bf16_rne(float f) {
    unsigned u = __float_as_uint(f);
    unsigned r = (u + 0x7FFFu + ((u >> 16) & 1u)) >> 16;
    return (unsigned short)r;
}

// ---------------------------------------------------------------------------
// CDNA5 async global -> LDS copy (16B per lane), ASYNCcnt-tracked.
// VDST = per-lane LDS byte address, VADDR = per-lane 64-bit global address.
// ---------------------------------------------------------------------------
__device__ __forceinline__ void async_copy_b128(unsigned lds_addr, const void* gaddr) {
    asm volatile("global_load_async_to_lds_b128 %0, %1, off"
                 :: "v"(lds_addr), "v"(gaddr)
                 : "memory");
}

__device__ __forceinline__ void wait_async0() {
    asm volatile("s_wait_asynccnt 0x0" ::: "memory");
}

// ---------------------------------------------------------------------------
// f32 -> bf16 conversion (flat)
// ---------------------------------------------------------------------------
__global__ void cvt_f32_bf16_kernel(const float* __restrict__ in,
                                    unsigned short* __restrict__ out,
                                    long long n) {
    long long i = (long long)blockIdx.x * blockDim.x + threadIdx.x;
    if (i < n) out[i] = f32_to_bf16_rne(in[i]);
}

// ---------------------------------------------------------------------------
// Weight transpose + convert: W[K,Nc] f32 (row-major) -> Wt[Nc,K] bf16
// ---------------------------------------------------------------------------
__global__ void cvt_w_transpose_kernel(const float* __restrict__ W,
                                       unsigned short* __restrict__ Wt,
                                       int K, int Nc) {
    int i = blockIdx.x * blockDim.x + threadIdx.x;
    if (i >= K * Nc) return;
    int k = i / Nc, n = i - k * Nc;
    Wt[(size_t)n * K + k] = f32_to_bf16_rne(W[i]);
}

// ---------------------------------------------------------------------------
// Fill
// ---------------------------------------------------------------------------
__global__ void fill_kernel(float* __restrict__ p, float v, long long n) {
    long long i = (long long)blockIdx.x * blockDim.x + threadIdx.x;
    if (i < n) p[i] = v;
}

// ---------------------------------------------------------------------------
// bf16 GEMM with WMMA:  C[M,Nc] = A[M,K] * Bt[Nc,K]^T,  f32 accumulate.
// Block tile 128x64, 8 waves; each wave owns a 16-row strip x 4 n-tiles.
// K staged in LDS (chunks of 32) with CDNA5 async global->LDS b128 copies.
// LDS strides padded (40 bf16 = 20 dwords) so the 16 lanes of a fragment
// read hit distinct bank groups.
// ---------------------------------------------------------------------------
#define TM 128
#define TN 64
#define TK 32
#define AS_LD 40   // padded stride (bf16 units), 80B: 16B-aligned, conflict-free
#define BS_LD 40

__device__ __forceinline__ v16bf load_frag_a(const unsigned short* As, int lane) {
    // 16-bit A 16x32: lanes 0-15: M=lane, K = 0..15
    //                 lanes16-31: M=lane-16, K = 16..31
    int m  = lane & 15;
    int kb = (lane >> 4) * 16;
    const unsigned short* p = As + m * AS_LD + kb;
    v8us lo = *(const v8us*)(p);        // K = kb .. kb+7
    v8us hi = *(const v8us*)(p + 8);    // K = kb+8 .. kb+15
    v16us t;
#pragma unroll
    for (int j = 0; j < 8; ++j) { t[j] = lo[j]; t[j + 8] = hi[j]; }
    return __builtin_bit_cast(v16bf, t);
}

__device__ __forceinline__ v16bf load_frag_b(const unsigned short* Bs, int lane, int col0) {
    // B 32x16: column = lane&15; lanes 0-15 K=0..15, lanes 16-31 K=16..31.
    // Bs is [n][k] so each lane's 16 elements are contiguous (32B).
    int n  = col0 + (lane & 15);
    int kb = (lane >> 4) * 16;
    return __builtin_bit_cast(v16bf, *(const v16us*)(Bs + n * BS_LD + kb));
}

__global__ __launch_bounds__(256)
void gemm_bf16_wmma_kernel(const unsigned short* __restrict__ A,
                           const unsigned short* __restrict__ Bt,
                           float* __restrict__ C,
                           int M, int K, int Nc) {
    __shared__ __align__(16) unsigned short As[TM * AS_LD]; // 10 KB
    __shared__ __align__(16) unsigned short Bs[TN * BS_LD]; // 5 KB

    const int tid  = threadIdx.x;
    const int wave = tid >> 5;
    const int lane = tid & 31;
    const int row0 = blockIdx.x * TM;
    const int col0 = blockIdx.y * TN;

    v8f acc[4] = {};

    for (int k0 = 0; k0 < K; k0 += TK) {
        // ---- stage A: 128 rows x 32 k. 512 async b128 copies (16B each).
        // OOB rows clamp to M-1 (their outputs are never stored).
#pragma unroll
        for (int it = 0; it < 2; ++it) {
            int idx = tid + it * 256;
            int r  = idx >> 2;
            int cc = (idx & 3) * 8;
            int gr = row0 + r;
            gr = gr < M ? gr : (M - 1);
            async_copy_b128((unsigned)(size_t)(As + r * AS_LD + cc),
                            A + (size_t)gr * K + k0 + cc);
        }
        // ---- stage B: 64 cols x 32 k from Wt[Nc,K]. 256 async b128 copies.
        {
            int n  = tid >> 2;           // 0..63
            int cc = (tid & 3) * 8;      // 0,8,16,24
            async_copy_b128((unsigned)(size_t)(Bs + n * BS_LD + cc),
                            Bt + (size_t)(col0 + n) * K + k0 + cc);
        }
        wait_async0();        // this wave's async copies are in LDS
        __syncthreads();      // all waves' copies visible

        v16bf a = load_frag_a(As + wave * 16 * AS_LD, lane);
#pragma unroll
        for (int t = 0; t < 4; ++t) {
            v16bf b = load_frag_b(Bs, lane, t * 16);
            acc[t] = __builtin_amdgcn_wmma_f32_16x16x32_bf16(
                false, a, false, b, (short)0, acc[t], false, false);
        }
        __syncthreads();      // fragments consumed; LDS reusable
    }

    // ---- store D: VGPR r -> row (r + 8*hi), col = lane&15 (ISA C/D layout)
    const int n  = lane & 15;
    const int mo = (lane >> 4) * 8;
#pragma unroll
    for (int t = 0; t < 4; ++t) {
#pragma unroll
        for (int r = 0; r < 8; ++r) {
            int gr = row0 + wave * 16 + mo + r;
            if (gr < M) C[(size_t)gr * Nc + col0 + t * 16 + n] = acc[t][r];
        }
    }
}

// ---------------------------------------------------------------------------
// Attention coefficients: a_src[n,h] = <h[n,h,:], att_src[h,:]>, same for dst.
// One wave per (node, head); lane covers channels {lane, lane+32}.
// ---------------------------------------------------------------------------
__global__ void att_coef_kernel(const float* __restrict__ h,
                                const float* __restrict__ att_src,
                                const float* __restrict__ att_dst,
                                float* __restrict__ a_src,
                                float* __restrict__ a_dst, int N) {
    int gw   = (int)(((long long)blockIdx.x * blockDim.x + threadIdx.x) >> 5);
    int lane = threadIdx.x & 31;
    if (gw >= N * HEADS) return;
    int n = gw >> 2, head = gw & 3;
    const float* hp  = h + (size_t)n * HC + head * CH;
    const float* asv = att_src + head * CH;
    const float* adv = att_dst + head * CH;
    float s = hp[lane] * asv[lane] + hp[lane + 32] * asv[lane + 32];
    float d = hp[lane] * adv[lane] + hp[lane + 32] * adv[lane + 32];
#pragma unroll
    for (int o = 16; o > 0; o >>= 1) {
        s += __shfl_xor(s, o, 32);
        d += __shfl_xor(d, o, 32);
    }
    if (lane == 0) { a_src[gw] = s; a_dst[gw] = d; }
}

__device__ __forceinline__ float leaky02(float v) { return v > 0.f ? v : 0.2f * v; }

// ---------------------------------------------------------------------------
// Edge pass 1: running max of leaky_relu(a_src[s]+a_dst[d]) per (dst, head)
// ---------------------------------------------------------------------------
__global__ void edge_max_kernel(const int* __restrict__ src, const int* __restrict__ dst,
                                const float* __restrict__ a_src, const float* __restrict__ a_dst,
                                float* __restrict__ mmax, int E) {
    long long t = (long long)blockIdx.x * blockDim.x + threadIdx.x;
    if (t >= (long long)E * HEADS) return;
    int e = (int)(t >> 2), head = (int)(t & 3);
    int s = src[e], d = dst[e];
    float v = leaky02(a_src[s * HEADS + head] + a_dst[d * HEADS + head]);
    atomicMax(&mmax[d * HEADS + head], v);
}

// ---------------------------------------------------------------------------
// Edge pass 2: w = exp(e - m[dst]);  denom[dst] += w;  store w per edge.
// ---------------------------------------------------------------------------
__global__ void edge_exp_kernel(const int* __restrict__ src, const int* __restrict__ dst,
                                const float* __restrict__ a_src, const float* __restrict__ a_dst,
                                const float* __restrict__ mmax,
                                float* __restrict__ denom, float* __restrict__ w_e, int E) {
    long long t = (long long)blockIdx.x * blockDim.x + threadIdx.x;
    if (t >= (long long)E * HEADS) return;
    int e = (int)(t >> 2), head = (int)(t & 3);
    int s = src[e], d = dst[e];
    float v = leaky02(a_src[s * HEADS + head] + a_dst[d * HEADS + head]);
    float w = __expf(v - mmax[d * HEADS + head]);
    w_e[t] = w;
    atomicAdd(&denom[d * HEADS + head], w);
}

// ---------------------------------------------------------------------------
// Scatter: agg[dst,h,c] += alpha * h[src,h,c]. Thread = (edge, head, ch-pair).
// ---------------------------------------------------------------------------
__global__ void scatter_kernel(const int* __restrict__ src, const int* __restrict__ dst,
                               const float* __restrict__ h,
                               const float* __restrict__ w_e, const float* __restrict__ denom,
                               float* __restrict__ agg, int E) {
    long long t = (long long)blockIdx.x * blockDim.x + threadIdx.x;
    if (t >= (long long)E * (HEADS * 32)) return;
    int lane = (int)(t & 31);
    int head = (int)((t >> 5) & 3);
    int e    = (int)(t >> 7);
    int s = src[e], d = dst[e];
    float alpha = w_e[(size_t)e * HEADS + head] /
                  (denom[d * HEADS + head] + 1e-16f);
    const float2 hv = *(const float2*)(h + (size_t)s * HC + head * CH + lane * 2);
    float* out = agg + (size_t)d * HC + head * CH + lane * 2;
    atomicAdd(out,     hv.x * alpha);
    atomicAdd(out + 1, hv.y * alpha);
}

// ---------------------------------------------------------------------------
// Layer-1 finalize: relu(agg + b1) -> bf16 (layer-2 GEMM input)
// ---------------------------------------------------------------------------
__global__ void finalize1_kernel(const float* __restrict__ agg, const float* __restrict__ bias,
                                 unsigned short* __restrict__ out_bf, long long total) {
    long long i = (long long)blockIdx.x * blockDim.x + threadIdx.x;
    if (i >= total) return;
    float v = agg[i] + bias[i & (HC - 1)];
    out_bf[i] = f32_to_bf16_rne(v > 0.f ? v : 0.f);
}

// ---------------------------------------------------------------------------
// Layer-2 finalize: relu(mean_h(agg) + b2), then row softmax over 64 channels.
// One wave per node; lane covers channels {lane, lane+32}.
// ---------------------------------------------------------------------------
__global__ void finalize2_softmax_kernel(const float* __restrict__ agg,
                                         const float* __restrict__ b2,
                                         float* __restrict__ out, int N) {
    int gw   = (int)(((long long)blockIdx.x * blockDim.x + threadIdx.x) >> 5);
    int lane = threadIdx.x & 31;
    if (gw >= N) return;
    const float* ap = agg + (size_t)gw * HC;
    float v0 = 0.f, v1 = 0.f;
#pragma unroll
    for (int h = 0; h < HEADS; ++h) {
        v0 += ap[h * CH + lane];
        v1 += ap[h * CH + lane + 32];
    }
    v0 = fmaxf(v0 * 0.25f + b2[lane], 0.f);
    v1 = fmaxf(v1 * 0.25f + b2[lane + 32], 0.f);
    float mx = fmaxf(v0, v1);
#pragma unroll
    for (int o = 16; o > 0; o >>= 1) mx = fmaxf(mx, __shfl_xor(mx, o, 32));
    float e0 = __expf(v0 - mx), e1 = __expf(v1 - mx);
    float s = e0 + e1;
#pragma unroll
    for (int o = 16; o > 0; o >>= 1) s += __shfl_xor(s, o, 32);
    float inv = 1.f / s;
    out[(size_t)gw * CH + lane]      = e0 * inv;
    out[(size_t)gw * CH + lane + 32] = e1 * inv;
}

// ---------------------------------------------------------------------------
// Host-side launch
// ---------------------------------------------------------------------------
extern "C" void kernel_launch(void* const* d_in, const int* in_sizes, int n_in,
                              void* d_out, int out_size, void* d_ws, size_t ws_size,
                              hipStream_t stream) {
    const float* x   = (const float*)d_in[0];
    const int*   ei  = (const int*)d_in[1];
    const float* W1  = (const float*)d_in[2];
    const float* as1 = (const float*)d_in[3];
    const float* ad1 = (const float*)d_in[4];
    const float* b1  = (const float*)d_in[5];
    const float* W2  = (const float*)d_in[6];
    const float* as2 = (const float*)d_in[7];
    const float* ad2 = (const float*)d_in[8];
    const float* b2  = (const float*)d_in[9];

    const int N = in_sizes[0] / 128;   // nodes
    const int E = in_sizes[1] / 2;     // edges (self-loops included)
    const int* src = ei;
    const int* dst = ei + E;

    // Workspace carve-up (256B aligned)
    char* ws = (char*)d_ws;
    size_t off = 0;
    auto carve = [&](size_t bytes) -> void* {
        void* p = ws + off;
        off = (off + bytes + 255) & ~(size_t)255;
        return p;
    };
    unsigned short* x_bf   = (unsigned short*)carve((size_t)N * 128 * 2);
    unsigned short* wt_bf  = (unsigned short*)carve((size_t)HC * HC * 2); // Wt[Nc][K]
    float*          h      = (float*)carve((size_t)N * HC * 4);
    unsigned short* h1_bf  = (unsigned short*)carve((size_t)N * HC * 2);
    float*          agg    = (float*)carve((size_t)N * HC * 4);
    float*          a_src  = (float*)carve((size_t)N * HEADS * 4);
    float*          a_dst  = (float*)carve((size_t)N * HEADS * 4);
    float*          mmax   = (float*)carve((size_t)N * HEADS * 4);
    float*          denom  = (float*)carve((size_t)N * HEADS * 4);
    float*          w_e    = (float*)carve((size_t)E * HEADS * 4);

    const int T = 256;
    auto blocks = [](long long n, int t) { return (unsigned)((n + t - 1) / t); };

    for (int layer = 0; layer < 2; ++layer) {
        const unsigned short* Abf;
        const float *asv, *adv;
        int K;
        if (layer == 0) {
            cvt_f32_bf16_kernel<<<blocks((long long)N * 128, T), T, 0, stream>>>(x, x_bf, (long long)N * 128);
            cvt_w_transpose_kernel<<<blocks(128 * HC, T), T, 0, stream>>>(W1, wt_bf, 128, HC);
            Abf = x_bf; K = 128; asv = as1; adv = ad1;
        } else {
            cvt_w_transpose_kernel<<<blocks(HC * HC, T), T, 0, stream>>>(W2, wt_bf, HC, HC);
            Abf = h1_bf; K = HC; asv = as2; adv = ad2;
        }

        dim3 gg((N + TM - 1) / TM, HC / TN);
        gemm_bf16_wmma_kernel<<<gg, T, 0, stream>>>(Abf, wt_bf, h, N, K, HC);

        att_coef_kernel<<<blocks((long long)N * HEADS * 32, T), T, 0, stream>>>(h, asv, adv, a_src, a_dst, N);

        fill_kernel<<<blocks((long long)N * HEADS, T), T, 0, stream>>>(mmax, -3.402823466e38f, (long long)N * HEADS);
        fill_kernel<<<blocks((long long)N * HEADS, T), T, 0, stream>>>(denom, 0.f, (long long)N * HEADS);
        fill_kernel<<<blocks((long long)N * HC, T), T, 0, stream>>>(agg, 0.f, (long long)N * HC);

        edge_max_kernel<<<blocks((long long)E * HEADS, T), T, 0, stream>>>(src, dst, a_src, a_dst, mmax, E);
        edge_exp_kernel<<<blocks((long long)E * HEADS, T), T, 0, stream>>>(src, dst, a_src, a_dst, mmax, denom, w_e, E);
        scatter_kernel<<<blocks((long long)E * HEADS * 32, T), T, 0, stream>>>(src, dst, h, w_e, denom, agg, E);

        if (layer == 0) {
            finalize1_kernel<<<blocks((long long)N * HC, T), T, 0, stream>>>(agg, b1, h1_bf, (long long)N * HC);
        } else {
            finalize2_softmax_kernel<<<blocks((long long)N * 32, T), T, 0, stream>>>(agg, b2, (float*)d_out, N);
        }
    }
}
